// TinySelfAttention_23338852287186
// MI455X (gfx1250) — matrix operand order repaired and checked by reading, not compiled, not run on hardware
//
#include <hip/hip_runtime.h>
#include <hip/hip_bf16.h>

// ---------------------------------------------------------------------------
// TinySelfAttention on MI455X (gfx1250): bf16 WMMA pipeline + TDM staging.
//   B=2, S=2048, D=1024, H=16, Hd=64.  Matrix-op bound (~54 GFLOP vs ~50MB).
// Stages (all on `stream`, scratch carved from d_ws):
//   1) cast x fp32 -> bf16
//   2) transpose-cast Wq/Wk/Wv/Wo -> bf16 K-major (WMMA B-fragment friendly)
//   3) GEMMs (TDM double-buffered LDS slabs, 8 wmma / 32-k-step / wave):
//        Q  = (x*WqT) * log2(e)/sqrt(Hd)   (bf16, row-major; scale pre-folded)
//        Kt = (x*WkT)^T                    (bf16, [D][B*S])
//        V  = x*WvT                        (bf16, row-major)
//   4) fused causal flash-attention, 64-key chunks:
//        scores 8 wmma + PV 8 wmma + ones-column 2 wmma (row sums for free),
//        rowmax via ds_swizzle XOR reductions, mask only on boundary chunks
//   5) GEMM out = y*WoT (fp32 out)
// ---------------------------------------------------------------------------

typedef __attribute__((ext_vector_type(16))) __bf16 v16bf;
typedef __attribute__((ext_vector_type(8)))  __bf16 v8bf;
typedef __attribute__((ext_vector_type(4)))  __bf16 v4bf;
typedef __attribute__((ext_vector_type(8)))  float  v8f;
typedef __attribute__((ext_vector_type(4)))  unsigned int u32x4;
typedef __attribute__((ext_vector_type(4)))  int  i32x4;
typedef __attribute__((ext_vector_type(8)))  int  i32x8;

constexpr int BATCH = 2;
constexpr int SEQ   = 2048;
constexpr int DIM   = 1024;
constexpr int HEADS = 16;
constexpr int HDIM  = 64;
constexpr int BS    = BATCH * SEQ;            // 4096 total rows
constexpr float SOFTMAX_SCALE_LOG2 = 0.18033688011112042f; // log2(e)/sqrt(64)

#if __has_builtin(__builtin_amdgcn_tensor_load_to_lds)
#define TDM_AVAILABLE 1
#else
#define TDM_AVAILABLE 0
#endif

__device__ __forceinline__ __bf16 f32_to_bf16(float f) {
  union { float f; unsigned u; } v; v.f = f;
  unsigned r = v.u + 0x7FFFu + ((v.u >> 16) & 1u);   // round-to-nearest-even
  unsigned short h = (unsigned short)(r >> 16);
  __bf16 b;
  __builtin_memcpy(&b, &h, 2);
  return b;
}

// Assemble a v16bf WMMA fragment from two contiguous 8-element chunks.
__device__ __forceinline__ v16bf load_frag16(const __bf16* p, int second_off_elems) {
  v8bf lo = *(const v8bf*)p;
  v8bf hi = *(const v8bf*)(p + second_off_elems);
  v16bf r;
#pragma unroll
  for (int i = 0; i < 8; ++i) { r[i] = lo[i]; r[i + 8] = hi[i]; }
  return r;
}

// ds_swizzle group-of-32 patterns (imm = xor<<10 | or<<5 | and).
template<int IMM>
__device__ __forceinline__ float swz(float x) {
  return __int_as_float(__builtin_amdgcn_ds_swizzle(__float_as_int(x), IMM));
}
constexpr int SWZ_X1   = (1 << 10) | 0x1F;
constexpr int SWZ_X2   = (2 << 10) | 0x1F;
constexpr int SWZ_X4   = (4 << 10) | 0x1F;
constexpr int SWZ_X8   = (8 << 10) | 0x1F;
constexpr int SWZ_BC16 = 0x10;   // new_lane = lane & 0x10 (bcast lane0/lane16)

// ---------------------------------------------------------------------------
// Tensor Data Mover: D# descriptors for a 2D bf16 tile (CDNA5 ISA).
// ---------------------------------------------------------------------------
#if TDM_AVAILABLE
__device__ __forceinline__ u32x4 tdm_group0(unsigned lds_off, const void* gptr) {
  unsigned long long ga = (unsigned long long)(size_t)gptr;
  u32x4 g = {};
  g[0] = 1u;                                                  // count=1, user mode
  g[1] = lds_off;                                             // lds_addr [63:32]
  g[2] = (unsigned)ga;                                        // global_addr lo
  g[3] = (unsigned)((ga >> 32) & 0x1FFFFFFull) | (2u << 30);  // addr hi + type=2
  return g;
}

// 2D tensor: dim0 contiguous (len td0, stride0 elems between rows), tile
// tile0 x tile1, data_size = 2 bytes, no multicast/pad/iterate.
__device__ __forceinline__ i32x8 tdm_group1_2d(unsigned td0, unsigned td1,
                                               unsigned tile0, unsigned tile1,
                                               unsigned stride0) {
  i32x8 g = {};
  g[0] = (int)(1u << 16);                                               // data_size=2B
  g[1] = (int)((td0 & 0xFFFFu) << 16);                                  // dim0 lo16
  g[2] = (int)(((td0 >> 16) & 0xFFFFu) | ((td1 & 0xFFFFu) << 16));      // dim0 hi/dim1 lo
  g[3] = (int)(((td1 >> 16) & 0xFFFFu) | ((tile0 & 0xFFFFu) << 16));    // dim1 hi/tile0
  g[4] = (int)(tile1 & 0xFFFFu);                                        // tile1, tile2=0
  g[5] = (int)stride0;                                                  // stride0 lo32
  g[6] = 0;
  g[7] = 0;
  return g;
}

__device__ __forceinline__ void tdm_load_2d(unsigned lds_off, const void* gptr,
                                            unsigned td0, unsigned td1,
                                            unsigned tile0, unsigned tile1,
                                            unsigned stride0) {
  u32x4 g0 = tdm_group0(lds_off, gptr);
  i32x8 g1 = tdm_group1_2d(td0, td1, tile0, tile1, stride0);
  i32x4 z4 = {};
#if defined(__clang_major__) && __clang_major__ >= 23
  i32x8 z8 = {};
  __builtin_amdgcn_tensor_load_to_lds(g0, g1, z4, z4, z8, 0);
#else
  __builtin_amdgcn_tensor_load_to_lds(g0, g1, z4, z4, 0);
#endif
}
#endif  // TDM_AVAILABLE

// ---------------------------------------------------------------------------
// Stage 1: elementwise fp32 -> bf16 cast (float4 in, v4bf out)
// ---------------------------------------------------------------------------
__global__ __launch_bounds__(256)
void cast_f32_bf16_x4(const float* __restrict__ src, __bf16* __restrict__ dst, int n4) {
  int i = blockIdx.x * 256 + threadIdx.x;
  if (i < n4) {
    float4 v = ((const float4*)src)[i];
    v4bf o = { f32_to_bf16(v.x), f32_to_bf16(v.y), f32_to_bf16(v.z), f32_to_bf16(v.w) };
    ((v4bf*)dst)[i] = o;
  }
}

// ---------------------------------------------------------------------------
// Stage 2: W [out][in] fp32 -> Wt [in][out] bf16 (LDS-tiled transpose)
// ---------------------------------------------------------------------------
__global__ __launch_bounds__(256)
void transpose_cast_w(const float* __restrict__ W, __bf16* __restrict__ Wt) {
  __shared__ float tile[32][33];
  const int i0 = blockIdx.x * 32;   // input-channel base (k)
  const int o0 = blockIdx.y * 32;   // output-channel base (n)
  const int tx = threadIdx.x & 31;
  const int ty = threadIdx.x >> 5;  // 0..7
#pragma unroll
  for (int r = 0; r < 32; r += 8)
    tile[r + ty][tx] = W[(size_t)(o0 + r + ty) * DIM + i0 + tx];
  __syncthreads();
#pragma unroll
  for (int r = 0; r < 32; r += 8)
    Wt[(size_t)(i0 + r + ty) * DIM + o0 + tx] = f32_to_bf16(tile[tx][r + ty]);
}

// ---------------------------------------------------------------------------
// Stage 3/5: C[M=4096][N=1024] = A[M][K=1024] * Wt[K][N], bf16 WMMA, f32 accum.
// Block = 128 rows x 64 cols (4 waves x 32 rows, shared 64-col Wt slab).
// A/Wt k-slabs staged in LDS via TDM (double-buffered) when available.
// OUT_MODE: 0 = bf16 row-major (*scale), 1 = bf16 transposed [N][BS], 2 = f32.
// ---------------------------------------------------------------------------
template<int OUT_MODE>
__global__ __launch_bounds__(128)
void gemm_xwt_wmma(const __bf16* __restrict__ A, const __bf16* __restrict__ Wt,
                   void* __restrict__ outp, float scale) {
  __shared__ __bf16 aslab[2][128][32];   // [buf][row][k]  16 KB
  __shared__ __bf16 wslab[2][32][64];    // [buf][k][n]     8 KB
  const int lane = threadIdx.x & 31;
  const int wave = threadIdx.x >> 5;
  const int bm   = blockIdx.x >> 4;      // 32 row-blocks of 128
  const int bn   = blockIdx.x & 15;      // 16 col-blocks of 64
  const int mRow0 = bm * 128;
  const int nBase = bn * 64;
  const int l16  = lane & 15;
  const int half = lane >> 4;

  v8f acc[2][4];
  {
    v8f z = {};
#pragma unroll
    for (int mt = 0; mt < 2; ++mt)
#pragma unroll
      for (int c = 0; c < 4; ++c) acc[mt][c] = z;
  }

#if TDM_AVAILABLE
  if (wave == 0) {
    tdm_load_2d((unsigned)(size_t)&aslab[0][0][0], A + (size_t)mRow0 * DIM,
                DIM, BS, 32, 128, DIM);
    tdm_load_2d((unsigned)(size_t)&wslab[0][0][0], Wt + nBase,
                DIM, DIM, 64, 32, DIM);
  }
#endif

  for (int kb = 0; kb < DIM; kb += 32) {
    const int cur = (kb >> 5) & 1;
#if TDM_AVAILABLE
    if (wave == 0) {
      if (kb + 32 < DIM) {
        const int nxt = cur ^ 1;
        tdm_load_2d((unsigned)(size_t)&aslab[nxt][0][0],
                    A + (size_t)mRow0 * DIM + (kb + 32), DIM, BS, 32, 128, DIM);
        tdm_load_2d((unsigned)(size_t)&wslab[nxt][0][0],
                    Wt + (size_t)(kb + 32) * DIM + nBase, DIM, DIM, 64, 32, DIM);
        __builtin_amdgcn_s_wait_tensorcnt(2);   // oldest pair (cur) complete
      } else {
        __builtin_amdgcn_s_wait_tensorcnt(0);
      }
    }
    __syncthreads();
#else
    __syncthreads();
    for (int i = threadIdx.x; i < 512; i += 128) {           // 128x32 A slab
      const int row = i >> 2, co = (i & 3) * 8;
      *(v8bf*)&aslab[cur][row][co] =
          *(const v8bf*)(A + (size_t)(mRow0 + row) * DIM + kb + co);
    }
    for (int i = threadIdx.x; i < 256; i += 128) {           // 32x64 Wt slab
      const int row = i >> 3, co = (i & 7) * 8;
      *(v8bf*)&wslab[cur][row][co] =
          *(const v8bf*)(Wt + (size_t)(kb + row) * DIM + nBase + co);
    }
    __syncthreads();
#endif

    v16bf a[2];
#pragma unroll
    for (int mt = 0; mt < 2; ++mt)
      a[mt] = load_frag16(&aslab[cur][wave * 32 + mt * 16 + l16][half * 8], 16);
#pragma unroll
    for (int c = 0; c < 4; ++c) {
      v16bf bfr;
#pragma unroll
      for (int s = 0; s < 16; ++s)
        bfr[s] = wslab[cur][half * 8 + (s & 7) + (s >> 3) * 16][c * 16 + l16];
#pragma unroll
      for (int mt = 0; mt < 2; ++mt)
        acc[mt][c] = __builtin_amdgcn_wmma_f32_16x16x32_bf16(
            false, a[mt], false, bfr, (short)0, acc[mt][c], false, false);
    }
    __syncthreads();   // all waves done reading before slab is overwritten
  }

#pragma unroll
  for (int mt = 0; mt < 2; ++mt)
#pragma unroll
    for (int c = 0; c < 4; ++c)
#pragma unroll
      for (int r = 0; r < 8; ++r) {
        const int row = mRow0 + wave * 32 + mt * 16 + r + half * 8;
        const int col = nBase + c * 16 + l16;
        const float v = acc[mt][c][r];
        if constexpr (OUT_MODE == 0)
          ((__bf16*)outp)[(size_t)row * DIM + col] = f32_to_bf16(v * scale);
        else if constexpr (OUT_MODE == 1)
          ((__bf16*)outp)[(size_t)col * BS + row] = f32_to_bf16(v * scale);
        else
          ((float*)outp)[(size_t)row * DIM + col] = v;
      }
}

// ---------------------------------------------------------------------------
// Stage 4: fused causal flash-attention (Q pre-scaled by log2e/sqrt(Hd)).
// ---------------------------------------------------------------------------
__global__ __launch_bounds__(128)
void attn_causal_wmma(const __bf16* __restrict__ Qb, const __bf16* __restrict__ Kt,
                      const __bf16* __restrict__ Vb, __bf16* __restrict__ Yb) {
  __shared__ __bf16 Plds[4][16][64];   // per-wave 2KB probability staging

  const int lane = threadIdx.x & 31;
  const int wave = threadIdx.x >> 5;
  const int l16  = lane & 15;
  const int half = lane >> 4;

  const int qblk = blockIdx.x & 31;          // S/64 = 32 query blocks
  const int bh   = blockIdx.x >> 5;
  const int b    = bh >> 4;
  const int h    = bh & 15;
  const int qBase = qblk * 64 + wave * 16;   // this wave's 16 query rows

  v16bf qf[2];
#pragma unroll
  for (int ks = 0; ks < 2; ++ks) {
    const __bf16* qp = Qb + (size_t)(b * SEQ + qBase + l16) * DIM
                          + h * HDIM + ks * 32 + half * 8;
    qf[ks] = load_frag16(qp, 16);
  }

  // B-fragment with ones in column 0 only: row sums of P via WMMA.
  v16bf onesb;
  {
    const __bf16 one = f32_to_bf16(1.0f), zer = f32_to_bf16(0.0f);
#pragma unroll
    for (int i = 0; i < 16; ++i) onesb[i] = (l16 == 0) ? one : zer;
  }

  v8f acc[4], lacc;
  {
    v8f z = {};
#pragma unroll
    for (int c = 0; c < 4; ++c) acc[c] = z;
    lacc = z;
  }
  float m[8];
#pragma unroll
  for (int r = 0; r < 8; ++r) m[r] = -1e30f;

  const int qMax = qBase + 15;
  for (int j0 = 0; j0 <= qMax; j0 += 64) {
    v8f sc[4];
    {
      v8f z = {};
#pragma unroll
      for (int t = 0; t < 4; ++t) sc[t] = z;
    }
#pragma unroll
    for (int t = 0; t < 4; ++t)
#pragma unroll
      for (int ks = 0; ks < 2; ++ks) {
        v16bf kf;
#pragma unroll
        for (int s = 0; s < 16; ++s)
          kf[s] = Kt[(size_t)(h * HDIM + ks * 32 + half * 8 + (s & 7) + (s >> 3) * 16) * BS
                     + b * SEQ + j0 + t * 16 + l16];
        sc[t] = __builtin_amdgcn_wmma_f32_16x16x32_bf16(
            false, qf[ks], false, kf, (short)0, sc[t], false, false);
      }

    if (j0 + 63 > qBase) {   // causal mask only on boundary chunks
#pragma unroll
      for (int t = 0; t < 4; ++t)
#pragma unroll
        for (int r = 0; r < 8; ++r) {
          const int key  = j0 + t * 16 + l16;
          const int qrow = qBase + r + half * 8;
          if (key > qrow) sc[t][r] = -1e30f;
        }
    }

    float al[8];
#pragma unroll
    for (int r = 0; r < 8; ++r) {
      float rm = fmaxf(fmaxf(sc[0][r], sc[1][r]), fmaxf(sc[2][r], sc[3][r]));
      rm = fmaxf(rm, swz<SWZ_X1>(rm));
      rm = fmaxf(rm, swz<SWZ_X2>(rm));
      rm = fmaxf(rm, swz<SWZ_X4>(rm));
      rm = fmaxf(rm, swz<SWZ_X8>(rm));
      const float mn = fmaxf(m[r], rm);
      al[r] = exp2f(m[r] - mn);
      m[r] = mn;
    }
#pragma unroll
    for (int t = 0; t < 4; ++t)
#pragma unroll
      for (int r = 0; r < 8; ++r)
        sc[t][r] = exp2f(sc[t][r] - m[r]);
#pragma unroll
    for (int c = 0; c < 4; ++c)
#pragma unroll
      for (int r = 0; r < 8; ++r) acc[c][r] *= al[r];
#pragma unroll
    for (int r = 0; r < 8; ++r) lacc[r] *= al[r];

#pragma unroll
    for (int t = 0; t < 4; ++t)
#pragma unroll
      for (int r = 0; r < 8; ++r)
        Plds[wave][r + half * 8][t * 16 + l16] = f32_to_bf16(sc[t][r]);
    v16bf pa[2];
#pragma unroll
    for (int ks = 0; ks < 2; ++ks)
      pa[ks] = load_frag16(&Plds[wave][l16][ks * 32 + half * 8], 16);

#pragma unroll
    for (int c = 0; c < 4; ++c)
#pragma unroll
      for (int ks = 0; ks < 2; ++ks) {
        v16bf vf;
#pragma unroll
        for (int s = 0; s < 16; ++s)
          vf[s] = Vb[(size_t)(b * SEQ + j0 + ks * 32 + half * 8 + (s & 7) + (s >> 3) * 16) * DIM
                     + h * HDIM + c * 16 + l16];
        acc[c] = __builtin_amdgcn_wmma_f32_16x16x32_bf16(
            false, pa[ks], false, vf, (short)0, acc[c], false, false);
      }
#pragma unroll
    for (int ks = 0; ks < 2; ++ks)
      lacc = __builtin_amdgcn_wmma_f32_16x16x32_bf16(
          false, pa[ks], false, onesb, (short)0, lacc, false, false);
  }

  float inv[8];
#pragma unroll
  for (int r = 0; r < 8; ++r) inv[r] = 1.0f / swz<SWZ_BC16>(lacc[r]);
#pragma unroll
  for (int c = 0; c < 4; ++c)
#pragma unroll
    for (int r = 0; r < 8; ++r) {
      const int row = qBase + r + half * 8;
      Yb[(size_t)(b * SEQ + row) * DIM + h * HDIM + c * 16 + l16] =
          f32_to_bf16(acc[c][r] * inv[r]);
    }
}

// ---------------------------------------------------------------------------
// Launcher
// ---------------------------------------------------------------------------
extern "C" void kernel_launch(void* const* d_in, const int* in_sizes, int n_in,
                              void* d_out, int out_size, void* d_ws, size_t ws_size,
                              hipStream_t stream) {
  const float* x  = (const float*)d_in[0];
  const float* Wq = (const float*)d_in[1];
  const float* Wk = (const float*)d_in[2];
  const float* Wv = (const float*)d_in[3];
  const float* Wo = (const float*)d_in[4];
  float* out = (float*)d_out;

  constexpr size_t X_ELEMS = (size_t)BS * DIM;        // 4,194,304
  constexpr size_t W_ELEMS = (size_t)DIM * DIM;       // 1,048,576
  constexpr size_t X_BYTES = X_ELEMS * 2;
  constexpr size_t W_BYTES = W_ELEMS * 2;

  char* ws = (char*)d_ws;
  __bf16* xb  = (__bf16*)(ws);
  __bf16* wqt = (__bf16*)(ws + X_BYTES);
  __bf16* wkt = (__bf16*)(ws + X_BYTES + 1 * W_BYTES);
  __bf16* wvt = (__bf16*)(ws + X_BYTES + 2 * W_BYTES);
  __bf16* wot = (__bf16*)(ws + X_BYTES + 3 * W_BYTES);
  __bf16* qb  = (__bf16*)(ws + X_BYTES + 4 * W_BYTES);
  __bf16* ktb = (__bf16*)(ws + 2 * X_BYTES + 4 * W_BYTES);  // [D][B*S]
  __bf16* vb  = (__bf16*)(ws + 3 * X_BYTES + 4 * W_BYTES);
  __bf16* yb  = (__bf16*)(ws + 4 * X_BYTES + 4 * W_BYTES);

  {
    const int n4 = (int)(X_ELEMS / 4);
    cast_f32_bf16_x4<<<(n4 + 255) / 256, 256, 0, stream>>>(x, xb, n4);
  }
  {
    dim3 g(DIM / 32, DIM / 32);
    transpose_cast_w<<<g, 256, 0, stream>>>(Wq, wqt);
    transpose_cast_w<<<g, 256, 0, stream>>>(Wk, wkt);
    transpose_cast_w<<<g, 256, 0, stream>>>(Wv, wvt);
    transpose_cast_w<<<g, 256, 0, stream>>>(Wo, wot);
  }
  const int gemm_blocks = (BS / 128) * (DIM / 64);    // 512
  gemm_xwt_wmma<0><<<gemm_blocks, 128, 0, stream>>>(xb, wqt, qb, SOFTMAX_SCALE_LOG2);
  gemm_xwt_wmma<1><<<gemm_blocks, 128, 0, stream>>>(xb, wkt, ktb, 1.0f);
  gemm_xwt_wmma<0><<<gemm_blocks, 128, 0, stream>>>(xb, wvt, vb, 1.0f);
  attn_causal_wmma<<<BATCH * HEADS * (SEQ / 64), 128, 0, stream>>>(qb, ktb, vb, yb);
  gemm_xwt_wmma<2><<<gemm_blocks, 128, 0, stream>>>(yb, wot, out, 1.0f);
}
